// MoleculeVAE_89412629168539
// MI455X (gfx1250) — matrix operand order repaired and checked by reading, not compiled, not run on hardware
//
#include <hip/hip_runtime.h>

// MoleculeVAE on gfx1250: bf16 WMMA GRU scans, weight-stationary + pipelined.
// 256-thread blocks (8 waves = 2/SIMD -> ~512 VGPR/wave). Each wave owns two
// H-tiles; 40 of its 48 Whh fragments are register-resident (the n-gate of
// the second tile streams from WGP$-resident global to stay under the VGPR
// cap without scratch spills). enc Wih / out_W live in padded LDS. Embedding
// gather is software-pipelined; one barrier per recurrence step.
#define Vv   100
#define Ee   128
#define Hh   256
#define Ll   64
#define Bb   4096
#define Tt   128
#define H3   768
#define BM   32
#define VPAD 112
// padded LDS row pitches (pitch % 32 dwords == 4 -> rows spread across banks)
#define LDH  (Hh + 8)   // 264 bf16
#define LDE  (Ee + 8)   // 136 bf16
#define LDZ  (Ll + 8)   // 72  bf16

typedef __attribute__((ext_vector_type(16))) __bf16 bf16x16;
typedef __attribute__((ext_vector_type(8)))  float  f32x8;

union ABFrag { bf16x16 v; uint4 q[2]; };

__device__ __forceinline__ unsigned short f2bf(float f) {
  unsigned u = __float_as_uint(f);
  unsigned r = u + 0x7FFFu + ((u >> 16) & 1u);   // round-to-nearest-even
  return (unsigned short)(r >> 16);
}

// A fragment (16x32 bf16, M x K): lane = row (lane&15),
// 16 contiguous K at kbase + 16*(lane>=16).
__device__ __forceinline__ bf16x16 load_frag_a(const unsigned short* tile, int ld,
                                               int kbase, int lane) {
  const int row = lane & 15;
  const int off = (lane & 16) ? 16 : 0;
  const unsigned short* p = tile + row * ld + kbase + off;
  ABFrag f;
  f.q[0] = *(const uint4*)(p);
  f.q[1] = *(const uint4*)(p + 8);
  return f.v;
}

// B fragment (32x16 bf16, K x N) from weights stored [N][K] row-major (== B^T):
// lane owns column N = lane&15, 16 contiguous K at kbase + 16*(lane>=16).
__device__ __forceinline__ bf16x16 load_frag_b(const unsigned short* W, int ldk,
                                               int nbase, int kbase, int lane) {
  const int n = lane & 15;
  const int kk = kbase + ((lane & 16) ? 16 : 0);
  const unsigned short* p = W + (nbase + n) * ldk + kk;
  ABFrag f;
  f.q[0] = *(const uint4*)(p);
  f.q[1] = *(const uint4*)(p + 8);
  return f.v;
}

__device__ __forceinline__ f32x8 wmma_bf16(bf16x16 a, bf16x16 b, f32x8 c) {
  return __builtin_amdgcn_wmma_f32_16x16x32_bf16(false, a, false, b, (short)0, c,
                                                 false, false);
}

__device__ __forceinline__ float sigmoidf_(float x) {
  return 1.0f / (1.0f + __expf(-x));
}

// ---------------------------------------------------------------- converters
__global__ void cvt_bf16_kernel(const float* __restrict__ src,
                                unsigned short* __restrict__ dst, int n) {
  int i = blockIdx.x * 256 + threadIdx.x;
  if (i < n) dst[i] = f2bf(src[i]);
}

__global__ void pad_outw_kernel(const float* __restrict__ outW,
                                unsigned short* __restrict__ dst) {
  int i = blockIdx.x * 256 + threadIdx.x;
  if (i < VPAD * Hh) {
    int rr = i / Hh;
    dst[i] = (rr < Vv) ? f2bf(outW[i]) : (unsigned short)0;
  }
}

// ---------------------------------------------------------------- encoder GRU
__global__ __launch_bounds__(256) void encoder_kernel(
    const int* __restrict__ x,
    const unsigned short* __restrict__ embB,
    const unsigned short* __restrict__ WihB,   // [3H][E] bf16
    const unsigned short* __restrict__ WhhB,   // [3H][H] bf16
    const float* __restrict__ bih,
    const float* __restrict__ bhh,
    float* __restrict__ h_last) {
  extern __shared__ char smem[];
  unsigned short* wihL = (unsigned short*)smem;                    // [H3][LDE]
  unsigned short* hbf  = (unsigned short*)(smem + H3 * LDE * 2);   // [2][BM][LDH]
  float*          h32  = (float*)(smem + H3 * LDE * 2 + 2 * BM * LDH * 2); // [BM][Hh]
  unsigned short* embA = (unsigned short*)((char*)h32 + BM * Hh * 4);      // [2][BM][LDE]
  int*            xIdx = (int*)((char*)embA + 2 * BM * LDE * 2);           // [BM][Tt]

  const int tid  = threadIdx.x;
  const int lane = tid & 31;
  const int w    = tid >> 5;          // 0..7
  const int b0   = blockIdx.x * BM;

  // stage Wih -> padded LDS
  for (int i = tid; i < H3 * 16; i += 256) {
    const int r = i >> 4, s = i & 15;
    *(uint4*)(wihL + r * LDE + s * 8) = *(const uint4*)(WihB + r * Ee + s * 8);
  }
  // block's x slice -> LDS (contiguous [BM][Tt] ints)
  for (int i = tid; i < BM * Tt / 4; i += 256)
    ((uint4*)xIdx)[i] = ((const uint4*)(x + (size_t)b0 * Tt))[i];
  for (int i = tid; i < BM * Hh; i += 256) h32[i] = 0.0f;
  for (int i = tid; i < BM * LDH; i += 256) hbf[i] = 0;  // buffer 0

  // register-resident Whh fragments: 40 frags (tile1's n-gate streams from
  // WGP$-cached global instead, keeping us under the VGPR cap -> no spills)
  bf16x16 wr[2][8], wz[2][8], wn0[8];
#pragma unroll
  for (int p = 0; p < 2; ++p) {
    const int jn = 2 * w + p;
#pragma unroll
    for (int k = 0; k < 8; ++k) {
      wr[p][k] = load_frag_b(WhhB, Hh, 0 * Hh + jn * 16, k * 32, lane);
      wz[p][k] = load_frag_b(WhhB, Hh, 1 * Hh + jn * 16, k * 32, lane);
    }
  }
#pragma unroll
  for (int k = 0; k < 8; ++k)
    wn0[k] = load_frag_b(WhhB, Hh, 2 * Hh + (2 * w) * 16, k * 32, lane);

  float br[2], bz[2], bxn[2], bhn[2];
#pragma unroll
  for (int p = 0; p < 2; ++p) {
    const int nc = (2 * w + p) * 16 + (lane & 15);
    br[p]  = bih[nc] + bhh[nc];
    bz[p]  = bih[Hh + nc] + bhh[Hh + nc];
    bxn[p] = bih[2 * Hh + nc];
    bhn[p] = bhh[2 * Hh + nc];
  }
  __syncthreads();   // xIdx/wihL/h ready

  // prologue: gather embeddings for t = 0 into buffer 0
  const int gi = tid >> 3, gseg = tid & 7;   // row 0..31, 16-bf16 segment 0..7
  {
    const int idx = xIdx[gi * Tt + 0];
    const uint4* src = (const uint4*)(embB + idx * Ee + gseg * 16);
    uint4* dst = (uint4*)(embA + gi * LDE + gseg * 16);
    dst[0] = src[0];
    dst[1] = src[1];
  }
  __syncthreads();

  int cur = 0;
#pragma unroll 1
  for (int t = 0; t < Tt; ++t) {
    // issue next step's embedding row loads early (latency hidden by WMMAs)
    uint4 e0, e1;
    if (t + 1 < Tt) {
      const int idx = xIdx[gi * Tt + (t + 1)];
      const uint4* src = (const uint4*)(embB + idx * Ee + gseg * 16);
      e0 = src[0];
      e1 = src[1];
    }

    const unsigned short* aEbase = embA + (t & 1) * BM * LDE;
    unsigned short* hNew = hbf + (cur ^ 1) * BM * LDH;
#pragma unroll
    for (int p = 0; p < 2; ++p) {
      const int jn = 2 * w + p;
      const int nc = jn * 16 + (lane & 15);
#pragma unroll
      for (int mt = 0; mt < 2; ++mt) {
        f32x8 ar, az, axn, ahn;
#pragma unroll
        for (int r = 0; r < 8; ++r) {
          ar[r] = br[p]; az[r] = bz[p]; axn[r] = bxn[p]; ahn[r] = bhn[p];
        }

        const unsigned short* aE = aEbase + mt * 16 * LDE;
#pragma unroll
        for (int k = 0; k < 4; ++k) {  // x-projection: A + B(Wih) from LDS
          bf16x16 a = load_frag_a(aE, LDE, k * 32, lane);
          ar  = wmma_bf16(a, load_frag_b(wihL, LDE, 0 * Hh + jn * 16, k * 32, lane), ar);
          az  = wmma_bf16(a, load_frag_b(wihL, LDE, 1 * Hh + jn * 16, k * 32, lane), az);
          axn = wmma_bf16(a, load_frag_b(wihL, LDE, 2 * Hh + jn * 16, k * 32, lane), axn);
        }
        const unsigned short* aH = hbf + cur * BM * LDH + mt * 16 * LDH;
#pragma unroll
        for (int k = 0; k < 8; ++k) {  // recurrent: B(Whh) from registers
          bf16x16 a = load_frag_a(aH, LDH, k * 32, lane);
          ar = wmma_bf16(a, wr[p][k], ar);
          az = wmma_bf16(a, wz[p][k], az);
          if (p == 0) {
            ahn = wmma_bf16(a, wn0[k], ahn);
          } else {  // tile1 n-gate: 8KB/wave slice, WGP$-resident after step 0
            ahn = wmma_bf16(a, load_frag_b(WhhB, Hh, 2 * Hh + jn * 16, k * 32, lane), ahn);
          }
        }

        const int m0 = mt * 16 + ((lane >> 4) << 3);
#pragma unroll
        for (int r = 0; r < 8; ++r) {
          const int m = m0 + r;
          const float hold = h32[m * Hh + nc];
          const float rr = sigmoidf_(ar[r]);
          const float zz = sigmoidf_(az[r]);
          const float nn = tanhf(axn[r] + rr * ahn[r]);
          const float hv = (1.0f - zz) * nn + zz * hold;
          h32[m * Hh + nc]   = hv;
          hNew[m * LDH + nc] = f2bf(hv);
        }
      }
    }

    if (t + 1 < Tt) {  // land prefetched rows into the other embA buffer
      uint4* dst = (uint4*)(embA + ((t + 1) & 1) * BM * LDE + gi * LDE + gseg * 16);
      dst[0] = e0;
      dst[1] = e1;
    }
    __syncthreads();   // single barrier per step
    cur ^= 1;
  }

  for (int i = tid; i < BM * Hh; i += 256) h_last[b0 * Hh + i] = h32[i];
}

// -------------------------------------------------------- latent (fused head)
__global__ __launch_bounds__(256) void latent_kernel(
    const float* __restrict__ h_last, const float* __restrict__ eps,
    const unsigned short* __restrict__ muWB, const float* __restrict__ mu_b,
    const unsigned short* __restrict__ lvWB, const float* __restrict__ lv_b,
    const unsigned short* __restrict__ diWB, const float* __restrict__ di_b,
    const unsigned short* __restrict__ decWihB, const float* __restrict__ dec_bih,
    float* __restrict__ out_mu, float* __restrict__ out_lv,
    float* __restrict__ hidden0, float* __restrict__ zp) {
  extern __shared__ char smem[];
  unsigned short* hA = (unsigned short*)smem;                    // [BM][LDH]
  unsigned short* zA = (unsigned short*)(smem + BM * LDH * 2);   // [BM][LDZ]

  const int tid  = threadIdx.x;
  const int lane = tid & 31;
  const int w    = tid >> 5;
  const int b0   = blockIdx.x * BM;

  for (int i = tid; i < BM * Hh; i += 256) {
    const int m = i >> 8, c = i & 255;
    hA[m * LDH + c] = f2bf(h_last[b0 * Hh + i]);
  }
  __syncthreads();

  {  // mu & logvar tiles: wave w -> (mt = w&1, lt = w>>1)
    const int mt = w & 1, lt = w >> 1;
    const int colL = lt * 16 + (lane & 15);
    f32x8 amu, alv;
    const float bm = mu_b[colL], bl = lv_b[colL];
#pragma unroll
    for (int r = 0; r < 8; ++r) { amu[r] = bm; alv[r] = bl; }
    const unsigned short* aH = hA + (mt * 16) * LDH;
#pragma unroll
    for (int k = 0; k < Hh; k += 32) {
      bf16x16 a = load_frag_a(aH, LDH, k, lane);
      amu = wmma_bf16(a, load_frag_b(muWB, Hh, lt * 16, k, lane), amu);
      alv = wmma_bf16(a, load_frag_b(lvWB, Hh, lt * 16, k, lane), alv);
    }
    const int m0 = mt * 16 + ((lane >> 4) << 3);
#pragma unroll
    for (int r = 0; r < 8; ++r) {
      const int m = m0 + r;
      const float mu = amu[r], lv = alv[r];
      out_mu[(b0 + m) * Ll + colL] = mu;
      out_lv[(b0 + m) * Ll + colL] = lv;
      const float z = mu + eps[(b0 + m) * Ll + colL] * __expf(0.5f * lv);
      zA[m * LDZ + colL] = f2bf(z);
    }
  }
  __syncthreads();

  for (int j = w; j < 32; j += 8) {  // hidden0 = tanh(z @ di_W^T + di_b)
    const int mt = j & 1, nt = j >> 1;
    const int col = nt * 16 + (lane & 15);
    f32x8 acc;
    const float bb = di_b[col];
#pragma unroll
    for (int r = 0; r < 8; ++r) acc[r] = bb;
    const unsigned short* aZ = zA + (mt * 16) * LDZ;
#pragma unroll
    for (int k = 0; k < Ll; k += 32) {
      bf16x16 a = load_frag_a(aZ, LDZ, k, lane);
      acc = wmma_bf16(a, load_frag_b(diWB, Ll, nt * 16, k, lane), acc);
    }
    const int m0 = mt * 16 + ((lane >> 4) << 3);
#pragma unroll
    for (int r = 0; r < 8; ++r)
      hidden0[(b0 + m0 + r) * Hh + col] = tanhf(acc[r]);
  }

  for (int j = w; j < 96; j += 8) {  // zp = z @ dec_Wih^T + dec_bih
    const int mt = j & 1, nt = j >> 1;
    const int col = nt * 16 + (lane & 15);
    f32x8 acc;
    const float bb = dec_bih[col];
#pragma unroll
    for (int r = 0; r < 8; ++r) acc[r] = bb;
    const unsigned short* aZ = zA + (mt * 16) * LDZ;
#pragma unroll
    for (int k = 0; k < Ll; k += 32) {
      bf16x16 a = load_frag_a(aZ, LDZ, k, lane);
      acc = wmma_bf16(a, load_frag_b(decWihB, Ll, nt * 16, k, lane), acc);
    }
    const int m0 = mt * 16 + ((lane >> 4) << 3);
#pragma unroll
    for (int r = 0; r < 8; ++r)
      zp[(b0 + m0 + r) * H3 + col] = acc[r];
  }
}

// -------------------------------------------- decoder GRU + fused logits GEMM
__global__ __launch_bounds__(256) void decoder_kernel(
    const unsigned short* __restrict__ WhhB,   // dec_Whh [3H][H] bf16
    const float* __restrict__ bhh,             // dec_bhh
    const float* __restrict__ hidden0, const float* __restrict__ zp,
    const unsigned short* __restrict__ outWB,  // [VPAD][Hh] bf16 (zero-padded)
    const float* __restrict__ out_b,
    float* __restrict__ logits) {
  extern __shared__ char smem[];
  float*          zpS   = (float*)smem;                                   // [BM][H3]
  float*          h32   = (float*)(smem + BM * H3 * 4);                   // [BM][Hh]
  unsigned short* hbf   = (unsigned short*)(smem + BM * H3 * 4 + BM * Hh * 4); // [2][BM][LDH]
  unsigned short* outWS = (unsigned short*)(smem + BM * H3 * 4 + BM * Hh * 4 +
                                            2 * BM * LDH * 2);            // [VPAD][LDH]

  const int tid  = threadIdx.x;
  const int lane = tid & 31;
  const int w    = tid >> 5;          // 0..7
  const int b0   = blockIdx.x * BM;

  for (int i = tid; i < BM * H3; i += 256) zpS[i] = zp[b0 * H3 + i];
  for (int i = tid; i < BM * Hh; i += 256) {
    const int m = i >> 8, c = i & 255;
    const float hv = hidden0[b0 * Hh + i];
    h32[i] = hv;
    hbf[m * LDH + c] = f2bf(hv);
  }
  for (int i = tid; i < VPAD * 32; i += 256) {  // 32 chunks of 8 bf16 per row
    const int r = i >> 5, s = i & 31;
    *(uint4*)(outWS + r * LDH + s * 8) = *(const uint4*)(outWB + r * Hh + s * 8);
  }

  bf16x16 wr[2][8], wz[2][8], wn[2][8];
#pragma unroll
  for (int p = 0; p < 2; ++p) {
    const int jn = 2 * w + p;
#pragma unroll
    for (int k = 0; k < 8; ++k) {
      wr[p][k] = load_frag_b(WhhB, Hh, 0 * Hh + jn * 16, k * 32, lane);
      wz[p][k] = load_frag_b(WhhB, Hh, 1 * Hh + jn * 16, k * 32, lane);
      wn[p][k] = load_frag_b(WhhB, Hh, 2 * Hh + jn * 16, k * 32, lane);
    }
  }
  float bhr[2], bhz[2], bhn[2];
#pragma unroll
  for (int p = 0; p < 2; ++p) {
    const int nc = (2 * w + p) * 16 + (lane & 15);
    bhr[p] = bhh[nc];
    bhz[p] = bhh[Hh + nc];
    bhn[p] = bhh[2 * Hh + nc];
  }
  __syncthreads();

  int cur = 0;
#pragma unroll 1
  for (int t = 0; t < Tt; ++t) {
    unsigned short* hNew = hbf + (cur ^ 1) * BM * LDH;
#pragma unroll
    for (int p = 0; p < 2; ++p) {  // GRU step (x-proj = constant zp)
      const int jn = 2 * w + p;
      const int nc = jn * 16 + (lane & 15);
#pragma unroll
      for (int mt = 0; mt < 2; ++mt) {
        const int m0 = mt * 16 + ((lane >> 4) << 3);
        f32x8 ar, az, ahn;
#pragma unroll
        for (int r = 0; r < 8; ++r) {
          ar[r]  = zpS[(m0 + r) * H3 + nc] + bhr[p];
          az[r]  = zpS[(m0 + r) * H3 + Hh + nc] + bhz[p];
          ahn[r] = bhn[p];
        }
        const unsigned short* aH = hbf + cur * BM * LDH + mt * 16 * LDH;
#pragma unroll
        for (int k = 0; k < 8; ++k) {
          bf16x16 a = load_frag_a(aH, LDH, k * 32, lane);
          ar  = wmma_bf16(a, wr[p][k], ar);
          az  = wmma_bf16(a, wz[p][k], az);
          ahn = wmma_bf16(a, wn[p][k], ahn);
        }
#pragma unroll
        for (int r = 0; r < 8; ++r) {
          const int m = m0 + r;
          const float hold = h32[m * Hh + nc];
          const float rr = sigmoidf_(ar[r]);
          const float zz = sigmoidf_(az[r]);
          const float zn = zpS[m * H3 + 2 * Hh + nc];
          const float nn = tanhf(zn + rr * ahn[r]);
          const float hv = (1.0f - zz) * nn + zz * hold;
          h32[m * Hh + nc]   = hv;
          hNew[m * LDH + nc] = f2bf(hv);
        }
      }
    }
    __syncthreads();   // h_new visible; single barrier per step

    // logits[b,t,:] = h_new @ out_W^T + out_b ; 14 tile jobs over 8 waves.
    for (int j = w; j < 14; j += 8) {
      const int mt2  = j & 1;
      const int vt   = j >> 1;
      const int colv = vt * 16 + (lane & 15);
      f32x8 acc;
      const float bb = (colv < Vv) ? out_b[colv] : 0.0f;
#pragma unroll
      for (int r = 0; r < 8; ++r) acc[r] = bb;
      const unsigned short* aH2 = hbf + (cur ^ 1) * BM * LDH + mt2 * 16 * LDH;
#pragma unroll
      for (int k = 0; k < Hh; k += 32) {
        bf16x16 a = load_frag_a(aH2, LDH, k, lane);
        acc = wmma_bf16(a, load_frag_b(outWS, LDH, vt * 16, k, lane), acc);
      }
      if (colv < Vv) {
        const int m0 = mt2 * 16 + ((lane >> 4) << 3);
#pragma unroll
        for (int r = 0; r < 8; ++r)
          logits[((size_t)(b0 + m0 + r) * Tt + t) * Vv + colv] = acc[r];
      }
    }
    cur ^= 1;
  }
}

// ---------------------------------------------------------------------- host
extern "C" void kernel_launch(void* const* d_in, const int* in_sizes, int n_in,
                              void* d_out, int out_size, void* d_ws, size_t ws_size,
                              hipStream_t stream) {
  (void)in_sizes; (void)n_in; (void)out_size; (void)ws_size;
  const int*   x         = (const int*)d_in[0];
  const float* eps       = (const float*)d_in[1];
  const float* embedding = (const float*)d_in[2];
  const float* enc_Wih   = (const float*)d_in[3];
  const float* enc_Whh   = (const float*)d_in[4];
  const float* enc_bih   = (const float*)d_in[5];
  const float* enc_bhh   = (const float*)d_in[6];
  const float* mu_W      = (const float*)d_in[7];
  const float* mu_b      = (const float*)d_in[8];
  const float* lv_W      = (const float*)d_in[9];
  const float* lv_b      = (const float*)d_in[10];
  const float* di_W      = (const float*)d_in[11];
  const float* di_b      = (const float*)d_in[12];
  const float* dec_Wih   = (const float*)d_in[13];
  const float* dec_Whh   = (const float*)d_in[14];
  const float* dec_bih   = (const float*)d_in[15];
  const float* dec_bhh   = (const float*)d_in[16];
  const float* out_W     = (const float*)d_in[17];
  const float* out_b     = (const float*)d_in[18];

  float* logits = (float*)d_out;
  float* out_mu = logits + (size_t)Bb * Tt * Vv;
  float* out_lv = out_mu + (size_t)Bb * Ll;

  char* ws = (char*)d_ws;
  size_t off = 0;
  auto carve = [&](size_t bytes) -> char* {
    char* p = ws + off;
    off += (bytes + 255) & ~(size_t)255;
    return p;
  };
  unsigned short* embB    = (unsigned short*)carve((size_t)Vv * Ee * 2);
  unsigned short* encWihB = (unsigned short*)carve((size_t)H3 * Ee * 2);
  unsigned short* encWhhB = (unsigned short*)carve((size_t)H3 * Hh * 2);
  unsigned short* muWB    = (unsigned short*)carve((size_t)Ll * Hh * 2);
  unsigned short* lvWB    = (unsigned short*)carve((size_t)Ll * Hh * 2);
  unsigned short* diWB    = (unsigned short*)carve((size_t)Hh * Ll * 2);
  unsigned short* decWihB = (unsigned short*)carve((size_t)H3 * Ll * 2);
  unsigned short* decWhhB = (unsigned short*)carve((size_t)H3 * Hh * 2);
  unsigned short* outWB   = (unsigned short*)carve((size_t)VPAD * Hh * 2);
  float* h_last  = (float*)carve((size_t)Bb * Hh * 4);
  float* hidden0 = (float*)carve((size_t)Bb * Hh * 4);
  float* zpW     = (float*)carve((size_t)Bb * H3 * 4);

  auto cvt = [&](const float* s, unsigned short* d, int n) {
    cvt_bf16_kernel<<<(n + 255) / 256, 256, 0, stream>>>(s, d, n);
  };
  cvt(embedding, embB, Vv * Ee);
  cvt(enc_Wih, encWihB, H3 * Ee);
  cvt(enc_Whh, encWhhB, H3 * Hh);
  cvt(mu_W, muWB, Ll * Hh);
  cvt(lv_W, lvWB, Ll * Hh);
  cvt(di_W, diWB, Hh * Ll);
  cvt(dec_Wih, decWihB, H3 * Ll);
  cvt(dec_Whh, decWhhB, H3 * Hh);
  pad_outw_kernel<<<(VPAD * Hh + 255) / 256, 256, 0, stream>>>(out_W, outWB);

  const size_t enc_smem = (size_t)H3 * LDE * 2 + (size_t)2 * BM * LDH * 2 +
                          (size_t)BM * Hh * 4 + (size_t)2 * BM * LDE * 2 +
                          (size_t)BM * Tt * 4;                 // ~302 KB (<320KB)
  encoder_kernel<<<Bb / BM, 256, enc_smem, stream>>>(
      x, embB, encWihB, encWhhB, enc_bih, enc_bhh, h_last);

  const size_t lat_smem = (size_t)BM * LDH * 2 + (size_t)BM * LDZ * 2;  // ~21 KB
  latent_kernel<<<Bb / BM, 256, lat_smem, stream>>>(
      h_last, eps, muWB, mu_b, lvWB, lv_b, diWB, di_b, decWihB, dec_bih,
      out_mu, out_lv, hidden0, zpW);

  const size_t dec_smem = (size_t)BM * H3 * 4 + (size_t)BM * Hh * 4 +
                          (size_t)2 * BM * LDH * 2 + (size_t)VPAD * LDH * 2; // ~219 KB
  decoder_kernel<<<Bb / BM, 256, dec_smem, stream>>>(
      decWhhB, dec_bhh, hidden0, zpW, outWB, out_b, logits);
}
